// AnonTokyoEncoder_88923002896630
// MI455X (gfx1250) — matrix operand order repaired and checked by reading, not compiled, not run on hardware
//
#include <hip/hip_runtime.h>
#include <hip/hip_bf16.h>

#define BB 4
#define AANUM 128
#define TT 11
#define MMNUM 512
#define PPTS 20
#define DMODEL 256
#define NHEAD 8
#define KTOP 32
#define HDIM 32
#define NLAYER 6
#define NEGV -1.0e9f

typedef __attribute__((ext_vector_type(16))) _Float16 v16h;
typedef __attribute__((ext_vector_type(8)))  float    v8f;

// ---------------------------------------------------------------------------
// Pack W[Kd,Nd] (f32, row-major) into WMMA-B-fragment layout, zero-padded to Kp:
//   WP[((tileC*KB + kb)*32 + lane)*16 + e] = W[(kb*32 + (lane>>4)*16 + e)*Nd + tileC*16 + (lane&15)]
// so the GEMM B-fragment is one contiguous 32-byte load per lane.
// ---------------------------------------------------------------------------
__global__ void pack_w_kernel(const float* __restrict__ W,
                              _Float16* __restrict__ WP,
                              int Kd, int Kp, int Nd)
{
  int tid = blockIdx.x * blockDim.x + threadIdx.x;     // over Nd * Kp
  if (tid >= Nd * Kp) return;
  int e    = tid & 15;
  int lane = (tid >> 4) & 31;
  int rest = tid >> 9;
  int KB   = Kp >> 5;
  int kb   = rest % KB;
  int tileC = rest / KB;
  int n = tileC * 16 + (lane & 15);
  int k = kb * 32 + (lane >> 4) * 16 + e;
  WP[tid] = (k < Kd) ? (_Float16)W[(size_t)k * Nd + n] : (_Float16)0.f;
}

// ---------------------------------------------------------------------------
// GEMM: Y[Nrows,Nd] = act(X[Nrows,Kp] @ W + bias); Kp % 32 == 0, Nrows % 16 == 0.
// One wave per 16x16 tile; A via 4x float4 loads, B via packed v16h load.
// A fragment layout: lane half h, K runs [16h,16h+16) (matches packed B).
// ---------------------------------------------------------------------------
__global__ void gemm_wmma_fast(const float* __restrict__ X,
                               const _Float16* __restrict__ WP,
                               const float* __restrict__ bias,
                               float* __restrict__ Y,
                               int Nrows, int Kp, int Nd, int relu)
{
  int gwave = (blockIdx.x * blockDim.x + threadIdx.x) >> 5;
  int lane  = threadIdx.x & 31;
  int tilesN = Nd >> 4;
  int rowTiles = Nrows >> 4;
  int tileR = gwave / tilesN;
  int tileC = gwave - tileR * tilesN;
  if (tileR >= rowTiles) return;                 // uniform per wave
  int half = lane >> 4;
  int l16  = lane & 15;
  int m = tileR * 16 + l16;
  int n = tileC * 16 + l16;
  int KB = Kp >> 5;

  const float4* xrow = (const float4*)(X + (size_t)m * Kp);
  const v16h*   wpv  = (const v16h*)WP + (size_t)tileC * KB * 32;

  v8f acc = {0.f, 0.f, 0.f, 0.f, 0.f, 0.f, 0.f, 0.f};
  for (int kb = 0; kb < KB; ++kb) {
    float4 a0 = xrow[kb * 8 + 4 * half + 0];
    float4 a1 = xrow[kb * 8 + 4 * half + 1];
    float4 a2 = xrow[kb * 8 + 4 * half + 2];
    float4 a3 = xrow[kb * 8 + 4 * half + 3];
    v16h bfrag = wpv[kb * 32 + lane];
    v16h afrag;
    afrag[0]  = (_Float16)a0.x; afrag[1]  = (_Float16)a0.y;
    afrag[2]  = (_Float16)a0.z; afrag[3]  = (_Float16)a0.w;
    afrag[4]  = (_Float16)a1.x; afrag[5]  = (_Float16)a1.y;
    afrag[6]  = (_Float16)a1.z; afrag[7]  = (_Float16)a1.w;
    afrag[8]  = (_Float16)a2.x; afrag[9]  = (_Float16)a2.y;
    afrag[10] = (_Float16)a2.z; afrag[11] = (_Float16)a2.w;
    afrag[12] = (_Float16)a3.x; afrag[13] = (_Float16)a3.y;
    afrag[14] = (_Float16)a3.z; afrag[15] = (_Float16)a3.w;
    acc = __builtin_amdgcn_wmma_f32_16x16x32_f16(false, afrag, false, bfrag,
                                                 (short)0, acc, false, false);
  }
  // C/D layout: VGPR r -> row = 8*half + r, col = lane&15
  float bv = bias ? bias[n] : 0.f;
  size_t obase = (size_t)(tileR * 16 + half * 8) * Nd + n;
#pragma unroll
  for (int r = 0; r < 8; ++r) {
    float v = acc[r] + bv;
    if (relu) v = v > 0.f ? v : 0.f;
    Y[obase + (size_t)r * Nd] = v;
  }
}

// ---------------------------------------------------------------------------
// Sparse attention: one wave per (b, token, head); lane = key slot (K=32),
// then lane = output channel (HD=32). q/k/v are (B,N,256) fp32.
// ---------------------------------------------------------------------------
__global__ void sparse_attn_kernel(const float* __restrict__ q,
                                   const float* __restrict__ kmat,
                                   const float* __restrict__ vmat,
                                   const int* __restrict__ idx,
                                   const float* __restrict__ kv_mask,
                                   float* __restrict__ out,
                                   int Nq, int Nkv)
{
  int wave = (blockIdx.x * blockDim.x + threadIdx.x) >> 5;
  int lane = threadIdx.x & 31;
  int total = BB * Nq * NHEAD;
  if (wave >= total) return;
  int h = wave % NHEAD;
  int n = (wave / NHEAD) % Nq;
  int b = wave / (NHEAD * Nq);

  int key = idx[((size_t)b * Nq + n) * KTOP + lane];
  float mk = kv_mask[b * Nkv + key];
  const float4* q4 = (const float4*)(q    + (((size_t)b * Nq  + n)   * NHEAD + h) * HDIM);
  const float4* k4 = (const float4*)(kmat + (((size_t)b * Nkv + key) * NHEAD + h) * HDIM);
  float s = 0.f;
#pragma unroll
  for (int j = 0; j < 8; ++j) {
    float4 qa = q4[j], ka = k4[j];
    s += qa.x * ka.x + qa.y * ka.y + qa.z * ka.z + qa.w * ka.w;
  }
  s *= 0.17677669529663687f;                     // 1/sqrt(32)
  s = (mk > 0.5f) ? s : NEGV;

  float mx = s;
  for (int off = 16; off > 0; off >>= 1) mx = fmaxf(mx, __shfl_xor(mx, off, 32));
  float e = expf(s - mx);
  float sum = e;
  for (int off = 16; off > 0; off >>= 1) sum += __shfl_xor(sum, off, 32);
  float a = e / sum;

  // o[lane] = sum_k a_k * v[key_k][h][lane]
  float o = 0.f;
#pragma unroll 4
  for (int kk = 0; kk < KTOP; ++kk) {
    float ak = __shfl(a, kk, 32);
    int   ik = __shfl(key, kk, 32);
    o += ak * vmat[(((size_t)b * Nkv + ik) * NHEAD + h) * HDIM + lane];
  }
  out[(((size_t)b * Nq + n) * NHEAD + h) * HDIM + lane] = o;
}

// ---------------------------------------------------------------------------
// out = LayerNorm(a + b) * g + beta, D=256, one wave per row (8 cols / lane)
// ---------------------------------------------------------------------------
__global__ void add_ln_kernel(const float* __restrict__ xa,
                              const float* __restrict__ xb,
                              const float* __restrict__ g,
                              const float* __restrict__ beta,
                              float* __restrict__ out, int Nrows)
{
  int wave = (blockIdx.x * blockDim.x + threadIdx.x) >> 5;
  int lane = threadIdx.x & 31;
  if (wave >= Nrows) return;
  size_t base = (size_t)wave * DMODEL;
  float x[8];
  float s = 0.f;
#pragma unroll
  for (int j = 0; j < 8; ++j) {
    int c = lane + 32 * j;
    x[j] = xa[base + c] + xb[base + c];
    s += x[j];
  }
  for (int off = 16; off > 0; off >>= 1) s += __shfl_xor(s, off, 32);
  float mean = s * (1.0f / DMODEL);
  float vs = 0.f;
#pragma unroll
  for (int j = 0; j < 8; ++j) { float d = x[j] - mean; vs += d * d; }
  for (int off = 16; off > 0; off >>= 1) vs += __shfl_xor(vs, off, 32);
  float inv = rsqrtf(vs * (1.0f / DMODEL) + 1e-5f);
#pragma unroll
  for (int j = 0; j < 8; ++j) {
    int c = lane + 32 * j;
    out[base + c] = (x[j] - mean) * inv * g[c] + beta[c];
  }
}

// ---------------------------------------------------------------------------
// PointNet helpers
// ---------------------------------------------------------------------------
__global__ void masked_pool_kernel(const float* __restrict__ h,
                                   const float* __restrict__ mask,
                                   float* __restrict__ pooled,
                                   int NG, int TP, int C)
{
  int tid = blockIdx.x * blockDim.x + threadIdx.x;
  if (tid >= NG * C) return;
  int gi = tid / C, c = tid - gi * C;
  float best = NEGV;
  for (int t = 0; t < TP; ++t) {
    float m = mask[gi * TP + t];
    float v = (m > 0.5f) ? h[((size_t)gi * TP + t) * C + c] : NEGV;
    best = fmaxf(best, v);
  }
  pooled[(size_t)gi * C + c] = best;
}

__global__ void concat_pool_kernel(const float* __restrict__ h,
                                   const float* __restrict__ pooled,
                                   float* __restrict__ hcat,
                                   int NG, int TP, int C)
{
  int tid = blockIdx.x * blockDim.x + threadIdx.x;
  if (tid >= NG * TP * C) return;
  int c = tid % C;
  int gt = tid / C;
  int gi = gt / TP;
  hcat[(size_t)gt * 2 * C + c]     = h[(size_t)gt * C + c];
  hcat[(size_t)gt * 2 * C + C + c] = pooled[(size_t)gi * C + c];
}

__global__ void zero_invalid_kernel(float* __restrict__ x,
                                    const float* __restrict__ mask,
                                    int NG, int TP, int C)
{
  int tid = blockIdx.x * blockDim.x + threadIdx.x;
  if (tid >= NG * C) return;
  int gi = tid / C;
  int valid = 0;
  for (int t = 0; t < TP; ++t) valid |= (mask[gi * TP + t] > 0.5f) ? 1 : 0;
  if (!valid) x[tid] = 0.f;
}

// ---------------------------------------------------------------------------
// Feature augmentation (rows padded to stride 32 so GEMM K-loop is guard-free)
// ---------------------------------------------------------------------------
__global__ void augment_agent_kernel(const float* __restrict__ trajs,
                                     const float* __restrict__ mask,
                                     float* __restrict__ out)
{
  int tid = blockIdx.x * blockDim.x + threadIdx.x;   // over B*A*T
  if (tid >= BB * AANUM * TT) return;
  int t = tid % TT;
  const float* tr = trajs + (size_t)tid * 10;
  float m = mask[tid];
  int mb = (m > 0.5f) ? 1 : 0;
  float o[29];
#pragma unroll
  for (int i = 0; i < 6; ++i) o[i] = tr[i];          // pos_size
#pragma unroll
  for (int i = 6; i < 11; ++i) o[i] = 0.f;           // type one-hot (zeros)
#pragma unroll
  for (int j = 0; j < TT; ++j) o[11 + j] = (j == t) ? 1.f : 0.f;  // eye(T)
  o[22] = (float)t * 0.1f;                           // linspace(0,1,11)
  o[23] = tr[6]; o[24] = tr[7];                      // heading
  o[25] = tr[8]; o[26] = tr[9];                      // vel
  int rowbase = tid - t;
  float vx, vy, pvx, pvy;
  if (t == 0) {  // acce[0] := acce[1] = (vel[1]-vel[0])/0.1
    vx  = trajs[(size_t)(rowbase + 1) * 10 + 8];
    vy  = trajs[(size_t)(rowbase + 1) * 10 + 9];
    pvx = trajs[(size_t)rowbase * 10 + 8];
    pvy = trajs[(size_t)rowbase * 10 + 9];
  } else {
    vx = tr[8]; vy = tr[9];
    pvx = trajs[(size_t)(tid - 1) * 10 + 8];
    pvy = trajs[(size_t)(tid - 1) * 10 + 9];
  }
  o[27] = (vx - pvx) * 10.f;
  o[28] = (vy - pvy) * 10.f;
#pragma unroll
  for (int i = 0; i < 29; ++i) out[(size_t)tid * 32 + i] = mb ? o[i] : 0.f;
  out[(size_t)tid * 32 + 29] = m;                    // raw mask channel
  out[(size_t)tid * 32 + 30] = 0.f;                  // K padding
  out[(size_t)tid * 32 + 31] = 0.f;
}

__global__ void augment_map_kernel(const float* __restrict__ polys,
                                   const float* __restrict__ mask,
                                   float* __restrict__ out)
{
  int tid = blockIdx.x * blockDim.x + threadIdx.x;   // over B*M*P
  if (tid >= BB * MMNUM * PPTS) return;
  int p = tid % PPTS;
  const float* mp = polys + (size_t)tid * 7;
  float m = mask[tid];
  int mb = (m > 0.5f) ? 1 : 0;
  int prow = (p == 0) ? tid : (tid - 1);             // pre[0] = xy[0]
#pragma unroll
  for (int i = 0; i < 7; ++i) out[(size_t)tid * 32 + i] = mb ? mp[i] : 0.f;
  out[(size_t)tid * 32 + 7] = mb ? polys[(size_t)prow * 7 + 0] : 0.f;
  out[(size_t)tid * 32 + 8] = mb ? polys[(size_t)prow * 7 + 1] : 0.f;
#pragma unroll
  for (int i = 9; i < 32; ++i) out[(size_t)tid * 32 + i] = 0.f;  // K padding
}

// ---------------------------------------------------------------------------
// Top-K nearest (K=32): one thread per query, running-worst selection
// ---------------------------------------------------------------------------
__global__ void topk_kernel(const float* __restrict__ qpos,
                            const float* __restrict__ kpos,
                            const float* __restrict__ kmask,
                            int* __restrict__ idxout, int Nq, int Nk)
{
  int tid = blockIdx.x * blockDim.x + threadIdx.x;   // over B*Nq
  if (tid >= BB * Nq) return;
  int b = tid / Nq;
  float qx = qpos[(size_t)tid * 2 + 0];
  float qy = qpos[(size_t)tid * 2 + 1];
  float bd[KTOP];
  int   bi[KTOP];
#pragma unroll
  for (int j = 0; j < KTOP; ++j) { bd[j] = 3.0e38f; bi[j] = 0; }
  int worst = 0;
  for (int kk = 0; kk < Nk; ++kk) {
    float m = kmask[b * Nk + kk];
    float dx = qx - kpos[((size_t)b * Nk + kk) * 2 + 0];
    float dy = qy - kpos[((size_t)b * Nk + kk) * 2 + 1];
    float d2 = (m > 0.5f) ? (dx * dx + dy * dy) : 3.0e38f;
    if (d2 < bd[worst]) {
      bd[worst] = d2; bi[worst] = kk;
      worst = 0;
      for (int j = 1; j < KTOP; ++j) if (bd[j] > bd[worst]) worst = j;
    }
  }
  for (int j = 0; j < KTOP; ++j) idxout[(size_t)tid * KTOP + j] = bi[j];
}

// ---------------------------------------------------------------------------
// Host orchestration
// ---------------------------------------------------------------------------
static void gemm_packed(const float* X, const float* W, const float* bias, float* Y,
                        int Nrows, int Kd, int Kp, int Nd, int relu,
                        _Float16* wp, hipStream_t s)
{
  int total = Nd * Kp;
  pack_w_kernel<<<(total + 255) / 256, 256, 0, s>>>(W, wp, Kd, Kp, Nd);
  int waves = (Nrows / 16) * (Nd / 16);
  gemm_wmma_fast<<<(waves * 32 + 255) / 256, 256, 0, s>>>(X, wp, bias, Y, Nrows, Kp, Nd, relu);
}

extern "C" void kernel_launch(void* const* d_in, const int* in_sizes, int n_in,
                              void* d_out, int out_size, void* d_ws, size_t ws_size,
                              hipStream_t stream)
{
  (void)in_sizes; (void)n_in; (void)out_size; (void)ws_size;
  auto Pf = [&](int i) -> const float* { return (const float*)d_in[i]; };

  const float* obj_trajs  = Pf(0);
  const float* obj_mask   = Pf(1);   // (B,A,T)
  const float* obj_pos    = Pf(2);   // (B,A,2)
  const float* agent_mask = Pf(4);   // (B,A)
  const float* map_polys  = Pf(5);   // (B,M,P,7)
  const float* poly_mask  = Pf(6);   // (B,M,P)
  const float* map_center = Pf(7);   // (B,M,2)
  const float* map_maskp  = Pf(9);   // (B,M)

  // param leaves (jax pytree sorted-key order):
  // agent_enc: 10 out.w 11 out.b 12 post0.w 13 post0.b 14 post1.w 15 post1.b
  //            16 pre0.w 17 pre0.b
  // layers[l]: base 18+48*l, blocks sorted (aa,am,mm), 16 leaves per block:
  //   +0 ffn1.w +1 ffn1.b +2 ffn2.w +3 ffn2.b +4 ffn_norm.g +5 ffn_norm.b
  //   +6 k.w +7 k.b +8 norm.g +9 norm.b +10 o.w +11 o.b +12 q.w +13 q.b +14 v.w +15 v.b
  // map_enc: 306 out.w 307 out.b 308 post0.w 309 post0.b 310 post1.w 311 post1.b
  //          312 pre0.w 313 pre0.b 314 pre1.w 315 pre1.b 316 pre2.w 317 pre2.b

  float* agent_feat = (float*)d_out;                             // (B,A,256)
  float* map_feat   = agent_feat + (size_t)BB * AANUM * DMODEL;  // (B,M,256)

  int* mm_idx = (int*)d_ws;
  int* aa_idx = mm_idx + (size_t)BB * MMNUM * KTOP;
  int* am_idx = aa_idx + (size_t)BB * AANUM * KTOP;
  _Float16* wptmp = (_Float16*)(am_idx + (size_t)BB * AANUM * KTOP); // 262144 halves max
  float* fws = (float*)(wptmp + 262144);

  const int NA = BB * AANUM * TT;     // 5632 agent points
  const int NM = BB * MMNUM * PPTS;   // 40960 map points
  const int RA = BB * AANUM;          // 512 agent tokens
  const int RM = BB * MMNUM;          // 2048 map tokens

  // ---------------- Agent PointNet ----------------
  {
    float* agent_in = fws;                                   // NA*32 (padded)
    float* hA    = agent_in + (size_t)NA * 32;               // NA*256
    float* poolA = hA + (size_t)NA * 256;                    // RA*256
    float* hcatA = poolA + (size_t)RA * 256;                 // NA*512
    float* h2A   = hcatA + (size_t)NA * 512;                 // NA*256
    float* h3A   = h2A + (size_t)NA * 256;                   // NA*256
    float* featA = h3A + (size_t)NA * 256;                   // RA*256

    augment_agent_kernel<<<(NA + 255) / 256, 256, 0, stream>>>(obj_trajs, obj_mask, agent_in);
    gemm_packed(agent_in, Pf(16), Pf(17), hA, NA, 30, 32, 256, 1, wptmp, stream);
    masked_pool_kernel<<<(RA * 256 + 255) / 256, 256, 0, stream>>>(hA, obj_mask, poolA, RA, TT, 256);
    concat_pool_kernel<<<(NA * 256 + 255) / 256, 256, 0, stream>>>(hA, poolA, hcatA, RA, TT, 256);
    gemm_packed(hcatA, Pf(12), Pf(13), h2A, NA, 512, 512, 256, 1, wptmp, stream);
    gemm_packed(h2A, Pf(14), Pf(15), h3A, NA, 256, 256, 256, 1, wptmp, stream);
    masked_pool_kernel<<<(RA * 256 + 255) / 256, 256, 0, stream>>>(h3A, obj_mask, featA, RA, TT, 256);
    gemm_packed(featA, Pf(10), Pf(11), agent_feat, RA, 256, 256, 256, 0, wptmp, stream);
    zero_invalid_kernel<<<(RA * 256 + 255) / 256, 256, 0, stream>>>(agent_feat, obj_mask, RA, TT, 256);
  }

  // ---------------- Map PointNet ----------------
  {
    float* mp32  = fws;                                      // NM*32 (padded)
    float* hMa   = mp32 + (size_t)NM * 32;                   // NM*64
    float* poolM = hMa + (size_t)NM * 64;                    // RM*64
    float* hcatM = poolM + (size_t)RM * 64;                  // NM*128
    float* hMb   = hcatM + (size_t)NM * 128;                 // NM*64
    float* featM = hMb + (size_t)NM * 64;                    // RM*64

    augment_map_kernel<<<(NM + 255) / 256, 256, 0, stream>>>(map_polys, poly_mask, mp32);
    gemm_packed(mp32, Pf(312), Pf(313), hMa, NM, 9, 32, 64, 1, wptmp, stream);
    gemm_packed(hMa, Pf(314), Pf(315), hMb, NM, 64, 64, 64, 1, wptmp, stream);
    gemm_packed(hMb, Pf(316), Pf(317), hMa, NM, 64, 64, 64, 1, wptmp, stream);
    masked_pool_kernel<<<(RM * 64 + 255) / 256, 256, 0, stream>>>(hMa, poly_mask, poolM, RM, PPTS, 64);
    concat_pool_kernel<<<(NM * 64 + 255) / 256, 256, 0, stream>>>(hMa, poolM, hcatM, RM, PPTS, 64);
    gemm_packed(hcatM, Pf(308), Pf(309), hMb, NM, 128, 128, 64, 1, wptmp, stream);
    gemm_packed(hMb, Pf(310), Pf(311), hMa, NM, 64, 64, 64, 1, wptmp, stream);
    masked_pool_kernel<<<(RM * 64 + 255) / 256, 256, 0, stream>>>(hMa, poly_mask, featM, RM, PPTS, 64);
    gemm_packed(featM, Pf(306), Pf(307), map_feat, RM, 64, 64, 256, 0, wptmp, stream);
    zero_invalid_kernel<<<(RM * 256 + 255) / 256, 256, 0, stream>>>(map_feat, poly_mask, RM, PPTS, 256);
  }

  // ---------------- Top-K neighbor tables ----------------
  topk_kernel<<<(BB * MMNUM + 255) / 256, 256, 0, stream>>>(map_center, map_center, map_maskp, mm_idx, MMNUM, MMNUM);
  topk_kernel<<<(BB * AANUM + 255) / 256, 256, 0, stream>>>(obj_pos, obj_pos, agent_mask, aa_idx, AANUM, AANUM);
  topk_kernel<<<(BB * AANUM + 255) / 256, 256, 0, stream>>>(obj_pos, map_center, map_maskp, am_idx, AANUM, MMNUM);

  // ---------------- Transformer layers ----------------
  float* bufQ = fws;                                         // RM*256 max
  float* bufK = bufQ + (size_t)RM * DMODEL;
  float* bufV = bufK + (size_t)RM * DMODEL;
  float* bufA = bufV + (size_t)RM * DMODEL;
  float* bufO = bufA + (size_t)RM * DMODEL;
  float* bufX = bufO + (size_t)RM * DMODEL;
  float* bufH = bufX + (size_t)RM * DMODEL;                  // RM*1024
  float* bufF = bufH + (size_t)RM * 1024;                    // RM*256

  auto run_block = [&](int lb, float* qf, const float* kvf, const float* kvmask,
                       const int* idx, int Nq, int Nkv) {
    int rq = BB * Nq, rkv = BB * Nkv;
    gemm_packed(qf,  Pf(lb + 12), Pf(lb + 13), bufQ, rq,  DMODEL, DMODEL, DMODEL, 0, wptmp, stream);
    gemm_packed(kvf, Pf(lb + 6),  Pf(lb + 7),  bufK, rkv, DMODEL, DMODEL, DMODEL, 0, wptmp, stream);
    gemm_packed(kvf, Pf(lb + 14), Pf(lb + 15), bufV, rkv, DMODEL, DMODEL, DMODEL, 0, wptmp, stream);
    {
      int th = BB * Nq * NHEAD * 32;
      sparse_attn_kernel<<<(th + 255) / 256, 256, 0, stream>>>(bufQ, bufK, bufV, idx, kvmask, bufA, Nq, Nkv);
    }
    gemm_packed(bufA, Pf(lb + 10), Pf(lb + 11), bufO, rq, DMODEL, DMODEL, DMODEL, 0, wptmp, stream);
    {
      int th = rq * 32;
      add_ln_kernel<<<(th + 255) / 256, 256, 0, stream>>>(qf, bufO, Pf(lb + 8), Pf(lb + 9), bufX, rq);
    }
    gemm_packed(bufX, Pf(lb + 0), Pf(lb + 1), bufH, rq, DMODEL, DMODEL, 4 * DMODEL, 1, wptmp, stream);
    gemm_packed(bufH, Pf(lb + 2), Pf(lb + 3), bufF, rq, 4 * DMODEL, 4 * DMODEL, DMODEL, 0, wptmp, stream);
    {
      int th = rq * 32;
      add_ln_kernel<<<(th + 255) / 256, 256, 0, stream>>>(bufX, bufF, Pf(lb + 4), Pf(lb + 5), qf, rq);
    }
  };

  for (int l = 0; l < NLAYER; ++l) {
    int base = 18 + 48 * l;   // leaves sorted: aa @ +0, am @ +16, mm @ +32
    run_block(base + 32, map_feat,   map_feat,   map_maskp,  mm_idx, MMNUM, MMNUM); // mm
    run_block(base + 0,  agent_feat, agent_feat, agent_mask, aa_idx, AANUM, AANUM); // aa
    run_block(base + 16, agent_feat, map_feat,   map_maskp,  am_idx, AANUM, MMNUM); // am
  }
}
